// MolAttentiveFP_EA_55508157333869
// MI455X (gfx1250) — matrix operand order repaired and checked by reading, not compiled, not run on hardware
//
#include <hip/hip_runtime.h>
#include <hip/hip_bf16.h>
#include <math.h>

// ---------------------------------------------------------------------------
// MolAttentiveFP on MI455X (gfx1250, wave32, WMMA)
//
// Dense math: node-level GEMMs on v_wmma_f32_16x16x32_f16 (f32 accumulate),
// f16 weights preconverted once, A strips staged through LDS, A fragments
// register-resident and reused across output column tiles.
// Edge-level attention reduced algebraically to scalar gathers + L2-resident
// scatter-adds (edge_attr == ones).
// ---------------------------------------------------------------------------

typedef __attribute__((ext_vector_type(16))) _Float16 v16h;
typedef __attribute__((ext_vector_type(8)))  _Float16 v8h;
typedef __attribute__((ext_vector_type(8)))  float    v8f;

#define N_NODES 50000
#define N_EDGES 800000
#define HID 128
#define H3  384
#define NG  1024

// ---------------------------------------------------------------------------
// f32 -> f16 weight conversion: dst[r*K+c] = src[r*srcld+c]
// ---------------------------------------------------------------------------
__global__ void k_cvt_f16(const float* __restrict__ src, _Float16* __restrict__ dst,
                          int rows, int srcld, int K)
{
    int i = blockIdx.x * blockDim.x + threadIdx.x;
    if (i >= rows * K) return;
    int r = i / K, c = i - r * K;
    dst[i] = (_Float16)src[(size_t)r * srcld + c];
}

// ---------------------------------------------------------------------------
// WMMA GEMM: C[M,Nout] = act(A[M,K] @ W16[Nout,K]^T + bias)
// grid.x = M/16, block = 128 (4 waves). One block owns a 16-row strip and all
// Nout/16 column tiles (wave w takes tiles w, w+4, ...).
// A strip f32->f16 into LDS once; per-wave A fragments preloaded into
// registers for all K/32 chunks and reused across column tiles.
// Fragment layouts per CDNA5 ISA 7.12.2:
//   A lane(0-15):row, halves K kb..kb+7 / kb+16..kb+23 (kb = 8*(lane>=16))
//   B lane: col = lane&15, halves K kbB..kbB+7 / kbB+16..kbB+23 (kbB = 8*(lane>=16))
// ACT: 0=none, 1=leaky(0.01), 2=relu  (compile-time -> branchless epilogue)
// ---------------------------------------------------------------------------
template <int K, int ACT>
__global__ __launch_bounds__(128) void wmma_gemm_t(const float* __restrict__ A,
                                                   const _Float16* __restrict__ W16,
                                                   const float* __restrict__ bias,
                                                   float* __restrict__ C,
                                                   int Nout)
{
    __shared__ _Float16 Atile[16 * K];

    const int tid = threadIdx.x;
    const size_t mbase = (size_t)blockIdx.x * 16;

    // cooperative f32->f16 stage of the A strip
    for (int i = tid; i < 16 * K; i += 128) {
        int r = i / K, c = i - r * K;
        Atile[i] = (_Float16)A[(mbase + r) * K + c];
    }
    __syncthreads();

    const int lane = tid & 31;
    const int wave = tid >> 5;
    const int half = lane >> 4;
    const int l15  = lane & 15;
    const int kbA  = half * 8;
    const int kbB  = half * 8;

    // preload this lane's A fragments for every K chunk (two b128 LDS reads each)
    v16h afrag[K / 32];
#pragma unroll
    for (int ck = 0; ck < K / 32; ++ck) {
        const _Float16* ap = &Atile[l15 * K + ck * 32 + kbA];
        v8h lo = *(const v8h*)ap;         // K = kb .. kb+7
        v8h hi = *(const v8h*)(ap + 16);  // K = kb+16 .. kb+23
        afrag[ck] = __builtin_shufflevector(lo, hi,
                     0, 1, 2, 3, 4, 5, 6, 7, 8, 9, 10, 11, 12, 13, 14, 15);
    }

    const int ntiles = Nout >> 4;
    for (int nt = wave; nt < ntiles; nt += 4) {
        const size_t colB = (size_t)nt * 16 + l15;
        const _Float16* brow = &W16[colB * K];
        if (nt + 4 < ntiles)  // prefetch next tile's B row (global_prefetch_b8)
            __builtin_prefetch(&W16[(colB + 64) * K], 0, 3);

        v8f c = {};
#pragma unroll
        for (int ck = 0; ck < K / 32; ++ck) {
            const _Float16* bp = brow + ck * 32 + kbB;
            v8h lo = *(const v8h*)bp;       // K = kbB .. kbB+7
            v8h hi = *(const v8h*)(bp + 16); // K = kbB+16 .. kbB+23
            v16h b = __builtin_shufflevector(lo, hi,
                      0, 1, 2, 3, 4, 5, 6, 7, 8, 9, 10, 11, 12, 13, 14, 15);
            c = __builtin_amdgcn_wmma_f32_16x16x32_f16(
                    /*neg_a=*/false, afrag[ck], /*neg_b=*/false, b,
                    /*c_mod=*/(short)0, c, /*reuse_a=*/false, /*reuse_b=*/false);
        }

        const size_t col  = (size_t)nt * 16 + l15;
        const size_t row0 = mbase + half * 8;
        const float  bv   = bias ? bias[col] : 0.f;   // invariant over r: hoisted
#pragma unroll
        for (int r = 0; r < 8; ++r) {
            float v = c[r] + bv;
            if (ACT == 1)      v = (v >= 0.f) ? v : 0.01f * v;
            else if (ACT == 2) v = fmaxf(v, 0.f);
            C[(row0 + r) * (size_t)Nout + col] = v;
        }
    }
}

// ---------------------------------------------------------------------------
// Small helpers
// ---------------------------------------------------------------------------
__device__ __forceinline__ float leakyf(float v) { return v >= 0.f ? v : 0.01f * v; }
__device__ __forceinline__ float sigmoidf_(float v) { return 1.f / (1.f + expf(-v)); }
// monotonic float<->uint key for atomicMax on floats
__device__ __forceinline__ unsigned ordKey(float f) {
    unsigned u = __float_as_uint(f);
    return (u & 0x80000000u) ? ~u : (u | 0x80000000u);
}
__device__ __forceinline__ float ordVal(unsigned u) {
    return (u & 0x80000000u) ? __uint_as_float(u & 0x7fffffffu) : __uint_as_float(~u);
}

__global__ void k_fill_f32(float* p, float v, size_t n) {
    size_t i = (size_t)blockIdx.x * blockDim.x + threadIdx.x;
    if (i < n) p[i] = v;
}
__global__ void k_fill_u32(unsigned* p, unsigned v, size_t n) {
    size_t i = (size_t)blockIdx.x * blockDim.x + threadIdx.x;
    if (i < n) p[i] = v;
}

// cedge[k] = sum_{j=128..152} Wg1[k][j]   (edge_attr == ones folded into bias)
__global__ void k_tail_rowsum(const float* __restrict__ Wg1, float* __restrict__ cedge) {
    int k = threadIdx.x;  // 128 threads
    float s = 0.f;
    for (int j = HID; j < HID + 25; ++j) s += Wg1[(size_t)k * 153 + j];
    cedge[k] = s;
}

// GATEConv node scalars: zs = dot(leaky(y+cedge), att_l), wd = dot(x, att_r)
__global__ void k_gate_scalars(const float* __restrict__ y, const float* __restrict__ cedge,
                               const float* __restrict__ att_l,
                               const float* __restrict__ x, const float* __restrict__ att_r,
                               float* __restrict__ zs, float* __restrict__ wd, int N)
{
    int n = blockIdx.x * blockDim.x + threadIdx.x;
    if (n >= N) return;
    float s0 = 0.f, s1 = 0.f;
    const size_t b = (size_t)n * HID;
    for (int c = 0; c < HID; ++c) {
        s0 += leakyf(y[b + c] + cedge[c]) * att_l[c];
        s1 += x[b + c] * att_r[c];
    }
    zs[n] = s0; wd[n] = s1;
}

// oa = xt . va ; ob = xt . vb (ob may be null)
__global__ void k_dual_dot(const float* __restrict__ xt,
                           const float* __restrict__ va, const float* __restrict__ vb,
                           float* __restrict__ oa, float* __restrict__ ob, int N)
{
    int n = blockIdx.x * blockDim.x + threadIdx.x;
    if (n >= N) return;
    float s0 = 0.f, s1 = 0.f;
    const size_t b = (size_t)n * HID;
    for (int c = 0; c < HID; ++c) {
        float v = xt[b + c];
        s0 += v * va[c];
        if (vb) s1 += v * vb[c];
    }
    oa[n] = s0;
    if (ob) ob[n] = s1;
}

// ---- edge-level segment softmax (segments = dst node) ----
__global__ void k_edge_logit_max(const int* __restrict__ src, const int* __restrict__ dst,
                                 const float* __restrict__ zs, const float* __restrict__ wd,
                                 float* __restrict__ elog, unsigned* __restrict__ maxb, int E)
{
    int e = blockIdx.x * blockDim.x + threadIdx.x;
    if (e >= E) return;
    float v = leakyf(zs[src[e]] + wd[dst[e]]);
    elog[e] = v;
    atomicMax(&maxb[dst[e]], ordKey(v));
}
__global__ void k_edge_exp_sum(const int* __restrict__ dst, float* __restrict__ elog,
                               const unsigned* __restrict__ maxb, float* __restrict__ sumb, int E)
{
    int e = blockIdx.x * blockDim.x + threadIdx.x;
    if (e >= E) return;
    float ex = expf(elog[e] - ordVal(maxb[dst[e]]));
    elog[e] = ex;
    atomicAdd(&sumb[dst[e]], ex);
}
__global__ void k_edge_scatter(const int* __restrict__ src, const int* __restrict__ dst,
                               const float* __restrict__ elog, const float* __restrict__ sumb,
                               const float* __restrict__ msg, float* __restrict__ acc, int E)
{
    size_t tid = (size_t)blockIdx.x * blockDim.x + threadIdx.x;
    if (tid >= (size_t)E * HID) return;
    int e = (int)(tid >> 7);
    int c = (int)(tid & 127);
    int d = dst[e];
    float coeff = elog[e] / (sumb[d] + 1e-16f);
    atomicAdd(&acc[(size_t)d * HID + c], msg[(size_t)src[e] * HID + c] * coeff);
}

// ---- node->graph segment softmax (segments = batch[n]) ----
__global__ void k_node_logit_max(const float* __restrict__ asrc, const float* __restrict__ ad,
                                 const int* __restrict__ batch, float* __restrict__ nlog,
                                 unsigned* __restrict__ maxb, int N)
{
    int n = blockIdx.x * blockDim.x + threadIdx.x;
    if (n >= N) return;
    float v = leakyf(asrc[n] + ad[batch[n]]);
    nlog[n] = v;
    atomicMax(&maxb[batch[n]], ordKey(v));
}
__global__ void k_node_exp_sum(const int* __restrict__ batch, float* __restrict__ nlog,
                               const unsigned* __restrict__ maxb, float* __restrict__ sumb, int N)
{
    int n = blockIdx.x * blockDim.x + threadIdx.x;
    if (n >= N) return;
    float ex = expf(nlog[n] - ordVal(maxb[batch[n]]));
    nlog[n] = ex;
    atomicAdd(&sumb[batch[n]], ex);
}
__global__ void k_node_scatter(const int* __restrict__ batch, const float* __restrict__ nlog,
                               const float* __restrict__ sumb, const float* __restrict__ xtm,
                               float* __restrict__ acc, int N)
{
    size_t tid = (size_t)blockIdx.x * blockDim.x + threadIdx.x;
    if (tid >= (size_t)N * HID) return;
    int n = (int)(tid >> 7);
    int c = (int)(tid & 127);
    int g = batch[n];
    float coeff = nlog[n] / (sumb[g] + 1e-16f);
    atomicAdd(&acc[(size_t)g * HID + c], xtm[tid] * coeff);
}

// h = elu(acc + bias[c])
__global__ void k_bias_elu(const float* __restrict__ acc, const float* __restrict__ bias,
                           float* __restrict__ h, size_t total)
{
    size_t i = (size_t)blockIdx.x * blockDim.x + threadIdx.x;
    if (i >= total) return;
    float v = acc[i] + bias[i & 127];
    h[i] = v > 0.f ? v : (expf(v) - 1.f);
}

// x = relu(gru(h, x)) given precomputed gates gi = h@Wih^T+bih, gh = x@Whh^T+bhh
__global__ void k_gru_combine(const float* __restrict__ gi, const float* __restrict__ gh,
                              float* __restrict__ x, size_t total)
{
    size_t i = (size_t)blockIdx.x * blockDim.x + threadIdx.x;
    if (i >= total) return;
    size_t n = i >> 7;
    int c = (int)(i & 127);
    const float* gin = gi + n * H3;
    const float* ghn = gh + n * H3;
    float r  = sigmoidf_(gin[c] + ghn[c]);
    float z  = sigmoidf_(gin[HID + c] + ghn[HID + c]);
    float nn = tanhf(gin[2 * HID + c] + r * ghn[2 * HID + c]);
    float o  = (1.f - z) * nn + z * x[i];
    x[i] = fmaxf(o, 0.f);
}

__global__ void k_readout_sum(const float* __restrict__ x, const int* __restrict__ batch,
                              float* __restrict__ o, size_t total)
{
    size_t i = (size_t)blockIdx.x * blockDim.x + threadIdx.x;
    if (i >= total) return;
    int n = (int)(i >> 7);
    int c = (int)(i & 127);
    atomicAdd(&o[(size_t)batch[n] * HID + c], x[i]);
}
__global__ void k_relu_inplace(float* p, size_t n) {
    size_t i = (size_t)blockIdx.x * blockDim.x + threadIdx.x;
    if (i < n) p[i] = fmaxf(p[i], 0.f);
}

// result[g] = o[g]·W2 + b2
__global__ void k_final(const float* __restrict__ o, const float* __restrict__ W2,
                        const float* __restrict__ b2, float* __restrict__ out)
{
    int g = blockIdx.x * blockDim.x + threadIdx.x;
    if (g >= NG) return;
    float s = b2[0];
    const size_t b = (size_t)g * HID;
    for (int c = 0; c < HID; ++c) s += o[b + c] * W2[c];
    out[g] = s;
}

// ---------------------------------------------------------------------------
extern "C" void kernel_launch(void* const* d_in, const int* in_sizes, int n_in,
                              void* d_out, int out_size, void* d_ws, size_t ws_size,
                              hipStream_t stream)
{
    (void)in_sizes; (void)n_in; (void)out_size; (void)ws_size;
    const int N = N_NODES, E = N_EDGES;

    const float* x_in   = (const float*)d_in[0];
    const int*   src    = (const int*)d_in[1];
    const int*   dst    = src + E;
    const int*   batch  = (const int*)d_in[2];
    const float* W1     = (const float*)d_in[3];
    const float* b1     = (const float*)d_in[4];
    const float* Wg1    = (const float*)d_in[5];
    const float* att_l  = (const float*)d_in[6];
    const float* att_r  = (const float*)d_in[7];
    const float* Wg2    = (const float*)d_in[8];
    const float* bg     = (const float*)d_in[9];
    const float* Wih0   = (const float*)d_in[10];
    const float* Whh0   = (const float*)d_in[11];
    const float* bih0   = (const float*)d_in[12];
    const float* bhh0   = (const float*)d_in[13];
    const float* Wa     = (const float*)d_in[14];
    const float* asrc_a = (const float*)d_in[15];
    const float* adst_a = (const float*)d_in[16];
    const float* ba     = (const float*)d_in[17];
    const float* Wih_a  = (const float*)d_in[18];
    const float* Whh_a  = (const float*)d_in[19];
    const float* bih_a  = (const float*)d_in[20];
    const float* bhh_a  = (const float*)d_in[21];
    const float* Wm     = (const float*)d_in[22];
    const float* asrc_m = (const float*)d_in[23];
    const float* adst_m = (const float*)d_in[24];
    const float* bm     = (const float*)d_in[25];
    const float* Wih_m  = (const float*)d_in[26];
    const float* Whh_m  = (const float*)d_in[27];
    const float* bih_m  = (const float*)d_in[28];
    const float* bhh_m  = (const float*)d_in[29];
    const float* W2     = (const float*)d_in[30];
    const float* b2     = (const float*)d_in[31];
    float* out = (float*)d_out;

    // ---- workspace layout ----
    float* ws = (float*)d_ws;
    size_t p = 0;
    float* xcur = ws + p; p += (size_t)N * HID;
    float* xt   = ws + p; p += (size_t)N * HID;   // y / xt / msg / xt_m
    float* acc  = ws + p; p += (size_t)N * HID;   // also acc2 (first NG*HID)
    float* hbuf = ws + p; p += (size_t)N * HID;
    float* gi   = ws + p; p += (size_t)N * H3;
    float* gh   = ws + p; p += (size_t)N * H3;
    float* zs   = ws + p; p += N;                 // also a_src_m
    float* wd   = ws + p; p += N;
    unsigned* maxb = (unsigned*)(ws + p); p += N;
    float* sumb = ws + p; p += N;
    float* elog = ws + p; p += E;                 // also nlog (N <= E)
    float* obuf = ws + p; p += (size_t)NG * HID;
    float* otb  = ws + p; p += (size_t)NG * HID;
    float* adb  = ws + p; p += NG;
    float* cedge = ws + p; p += HID;

    // f16 weight area (16-byte aligned: all offsets above are multiples of 16 floats)
    _Float16* hw = (_Float16*)(ws + p);
    size_t q = 0;
    _Float16* w1h   = hw + q; q += (size_t)128 * 64;
    _Float16* wg1h  = hw + q; q += (size_t)128 * 128;
    _Float16* wg2h  = hw + q; q += (size_t)128 * 128;
    _Float16* wih0h = hw + q; q += (size_t)H3 * 128;
    _Float16* whh0h = hw + q; q += (size_t)H3 * 128;
    _Float16* wah   = hw + q; q += (size_t)2 * 128 * 128;
    _Float16* wihah = hw + q; q += (size_t)2 * H3 * 128;
    _Float16* whhah = hw + q; q += (size_t)2 * H3 * 128;
    _Float16* wmh   = hw + q; q += (size_t)128 * 128;
    _Float16* wihmh = hw + q; q += (size_t)H3 * 128;
    _Float16* whhmh = hw + q; q += (size_t)H3 * 128;

    const dim3 B256(256);
    auto blks = [](size_t n) { return dim3((unsigned)((n + 255) / 256)); };

    auto cvt = [&](const float* s, _Float16* d, int rows, int srcld, int K) {
        k_cvt_f16<<<blks((size_t)rows * K), B256, 0, stream>>>(s, d, rows, srcld, K);
    };
    // act: 0=none, 1=leaky, 2=relu (dispatched to compile-time ACT)
    auto gemm = [&](const float* A, const _Float16* W16, const float* bias,
                    float* C, int M, int K, int Nout, int act) {
        dim3 g(M / 16), b(128);
        if (K == 64) {
            if (act == 1)      wmma_gemm_t<64, 1><<<g, b, 0, stream>>>(A, W16, bias, C, Nout);
            else if (act == 2) wmma_gemm_t<64, 2><<<g, b, 0, stream>>>(A, W16, bias, C, Nout);
            else               wmma_gemm_t<64, 0><<<g, b, 0, stream>>>(A, W16, bias, C, Nout);
        } else {
            if (act == 1)      wmma_gemm_t<128, 1><<<g, b, 0, stream>>>(A, W16, bias, C, Nout);
            else if (act == 2) wmma_gemm_t<128, 2><<<g, b, 0, stream>>>(A, W16, bias, C, Nout);
            else               wmma_gemm_t<128, 0><<<g, b, 0, stream>>>(A, W16, bias, C, Nout);
        }
    };

    auto edge_round = [&](const float* zsrc, const float* wdst, const float* msg) {
        k_fill_u32<<<blks(N), B256, 0, stream>>>(maxb, 0u, (size_t)N);
        k_fill_f32<<<blks(N), B256, 0, stream>>>(sumb, 0.f, (size_t)N);
        k_fill_f32<<<blks((size_t)N * HID), B256, 0, stream>>>(acc, 0.f, (size_t)N * HID);
        k_edge_logit_max<<<blks(E), B256, 0, stream>>>(src, dst, zsrc, wdst, elog, maxb, E);
        k_edge_exp_sum<<<blks(E), B256, 0, stream>>>(dst, elog, maxb, sumb, E);
        k_edge_scatter<<<blks((size_t)E * HID), B256, 0, stream>>>(src, dst, elog, sumb, msg, acc, E);
    };

    // ---- one-time weight conversions (tiny, L2-resident) ----
    cvt(W1, w1h, 128, 64, 64);
    cvt(Wg1, wg1h, 128, 153, 128);     // strip edge-feature tail columns
    cvt(Wg2, wg2h, 128, 128, 128);
    cvt(Wih0, wih0h, H3, 128, 128);
    cvt(Whh0, whh0h, H3, 128, 128);
    cvt(Wa, wah, 2 * 128, 128, 128);
    cvt(Wih_a, wihah, 2 * H3, 128, 128);
    cvt(Whh_a, whhah, 2 * H3, 128, 128);
    cvt(Wm, wmh, 128, 128, 128);
    cvt(Wih_m, wihmh, H3, 128, 128);
    cvt(Whh_m, whhmh, H3, 128, 128);

    // ---- layer 0: x = leaky(x @ W1^T + b1) ----
    gemm(x_in, w1h, b1, xcur, N, 64, HID, /*leaky*/1);

    // ---- GATEConv ----
    k_tail_rowsum<<<dim3(1), dim3(HID), 0, stream>>>(Wg1, cedge);
    gemm(xcur, wg1h, nullptr, xt, N, HID, HID, 0);                 // y = x @ Wg1a^T
    k_gate_scalars<<<blks(N), B256, 0, stream>>>(xt, cedge, att_l, xcur, att_r, zs, wd, N);
    gemm(xcur, wg2h, nullptr, xt, N, HID, HID, 0);                 // msg = x @ Wg2^T
    edge_round(zs, wd, xt);
    k_bias_elu<<<blks((size_t)N * HID), B256, 0, stream>>>(acc, bg, hbuf, (size_t)N * HID);
    gemm(hbuf, wih0h, bih0, gi, N, HID, H3, 0);
    gemm(xcur, whh0h, bhh0, gh, N, HID, H3, 0);
    k_gru_combine<<<blks((size_t)N * HID), B256, 0, stream>>>(gi, gh, xcur, (size_t)N * HID);

    // ---- 2 GATConv + GRU layers ----
    for (int l = 0; l < 2; ++l) {
        gemm(xcur, wah + (size_t)l * HID * HID, nullptr, xt, N, HID, HID, 0);
        k_dual_dot<<<blks(N), B256, 0, stream>>>(xt, asrc_a + l * HID, adst_a + l * HID, zs, wd, N);
        edge_round(zs, wd, xt);
        k_bias_elu<<<blks((size_t)N * HID), B256, 0, stream>>>(acc, ba + l * HID, hbuf, (size_t)N * HID);
        gemm(hbuf, wihah + (size_t)l * H3 * HID, bih_a + l * H3, gi, N, HID, H3, 0);
        gemm(xcur, whhah + (size_t)l * H3 * HID, bhh_a + l * H3, gh, N, HID, H3, 0);
        k_gru_combine<<<blks((size_t)N * HID), B256, 0, stream>>>(gi, gh, xcur, (size_t)N * HID);
    }

    // ---- attentive readout ----
    k_fill_f32<<<blks((size_t)NG * HID), B256, 0, stream>>>(obuf, 0.f, (size_t)NG * HID);
    k_readout_sum<<<blks((size_t)N * HID), B256, 0, stream>>>(xcur, batch, obuf, (size_t)N * HID);
    k_relu_inplace<<<blks((size_t)NG * HID), B256, 0, stream>>>(obuf, (size_t)NG * HID);

    gemm(xcur, wmh, nullptr, xt, N, HID, HID, 0);                  // xt_m (fixed)
    k_dual_dot<<<blks(N), B256, 0, stream>>>(xt, asrc_m, nullptr, zs, nullptr, N);  // a_src

    for (int t = 0; t < 3; ++t) {
        gemm(obuf, wmh, nullptr, otb, NG, HID, HID, 0);            // ot = out @ Wm^T
        k_dual_dot<<<blks(NG), B256, 0, stream>>>(otb, adst_m, nullptr, adb, nullptr, NG);
        k_fill_u32<<<blks(NG), B256, 0, stream>>>(maxb, 0u, (size_t)NG);
        k_fill_f32<<<blks(NG), B256, 0, stream>>>(sumb, 0.f, (size_t)NG);
        k_fill_f32<<<blks((size_t)NG * HID), B256, 0, stream>>>(acc, 0.f, (size_t)NG * HID);
        k_node_logit_max<<<blks(N), B256, 0, stream>>>(zs, adb, batch, elog, maxb, N);
        k_node_exp_sum<<<blks(N), B256, 0, stream>>>(batch, elog, maxb, sumb, N);
        k_node_scatter<<<blks((size_t)N * HID), B256, 0, stream>>>(batch, elog, sumb, xt, acc, N);
        k_bias_elu<<<blks((size_t)NG * HID), B256, 0, stream>>>(acc, bm, hbuf, (size_t)NG * HID);
        gemm(hbuf, wihmh, bih_m, gi, NG, HID, H3, 0);
        gemm(obuf, whhmh, bhh_m, gh, NG, HID, H3, 0);
        k_gru_combine<<<blks((size_t)NG * HID), B256, 0, stream>>>(gi, gh, obuf, (size_t)NG * HID);
    }

    k_final<<<dim3(NG / 256), B256, 0, stream>>>(obuf, W2, b2, out);
}
